// ScaledDotProductAttention_encoder_88656714924845
// MI455X (gfx1250) — matrix operand order repaired and checked by reading, not compiled, not run on hardware
//
#include <hip/hip_runtime.h>
#include <cstdint>

// Scaled dot product attention, B=8 H=8 S=1024 Dk=64, outputs (context, attn).
// Fused one-pass design: QK^T via bf16 WMMA -> scores in LDS -> softmax ->
// PV via bf16 WMMA. One workgroup (4 waves, 128 thr) per 16-query strip.

#define B_   8
#define H_   8
#define S_   1024
#define DK   64
#define QT   16            // query tile rows per workgroup
#define RS   1028          // LDS row stride in floats (pad: 4112B % 256B = 16 -> conflict-free)
#define NEGV (-1.0e9f)
#define SCALE 0.125f       // 1/sqrt(64)

typedef __attribute__((ext_vector_type(16))) __bf16 bf16x16;
typedef __attribute__((ext_vector_type(8)))  float  f32x8;

// A-fragment (16x32 bf16) element j -> K offset within the 32-wide chunk,
// given lane-half kh (lane>>4). Per ISA 7.12.2 16-bit A layout.
__device__ __forceinline__ int a_koff(int j) {
    return ((j >> 3) * 16) + (((j >> 1) & 3) * 2) + (j & 1);
}

__global__ void
__launch_bounds__(128, 1)
sdpa_wmma_kernel(const float* __restrict__ Q,
                 const float* __restrict__ K,
                 const float* __restrict__ V,
                 const uint8_t* __restrict__ mask,
                 const float* __restrict__ bias,
                 float* __restrict__ ctx_out,
                 float* __restrict__ attn_out) {
    extern __shared__ float smem[];           // [QT*RS] scores/probs, then stats
    float* probs  = smem;                     // QT x RS
    float* wmax   = smem + QT * RS;           // 4 waves x 16 rows
    float* rowsum = wmax + 4 * QT;            // 16

    const int tid  = threadIdx.x;
    const int lane = tid & 31;
    const int wave = tid >> 5;
    const int kh   = lane >> 4;               // lane half (0/1)
    const int ln16 = lane & 15;

    const int nqb = S_ / QT;                  // 64 query blocks per head
    const int bh  = blockIdx.x / nqb;         // fused (b,h)
    const int qb  = blockIdx.x % nqb;

    const size_t kvBase   = (size_t)bh * S_ * DK;
    const size_t qBase    = kvBase + (size_t)qb * QT * DK;
    const size_t maskBase = ((size_t)bh * S_ + (size_t)qb * QT) * S_;
    const size_t ctxBase  = qBase;

    // ---------------- Phase 1: Q tile -> two 16x32 bf16 A-fragments ----------
    bf16x16 a0, a1;
    {
        const float* qp = Q + qBase + (size_t)ln16 * DK + kh * 8;
#pragma unroll
        for (int j = 0; j < 16; ++j) {
            int off = a_koff(j);
            a0[j] = (__bf16)qp[off];          // K dims 0..31
            a1[j] = (__bf16)qp[off + 32];     // K dims 32..63
        }
    }

    // ---------------- Phase 2: scores = QK^T*scale + bias, mask --------------
    float mx[8];
#pragma unroll
    for (int r = 0; r < 8; ++r) mx[r] = -3.0e38f;

    for (int t = 0; t < 16; ++t) {
        const int kb = wave * 16 + t;         // this wave's key block (16 keys)

        // B-fragments (32x16 bf16): lane half h holds d 8h..8h+7 and 16+8h..23+8h per 32-d chunk.
        bf16x16 b0, b1;
        {
            const float* kp = K + kvBase + (size_t)(kb * 16 + ln16) * DK + kh * 8;
            float4 x0 = *(const float4*)(kp + 0);
            float4 x1 = *(const float4*)(kp + 4);
            float4 x2 = *(const float4*)(kp + 16);
            float4 x3 = *(const float4*)(kp + 20);
            float4 y0 = *(const float4*)(kp + 32);
            float4 y1 = *(const float4*)(kp + 36);
            float4 y2 = *(const float4*)(kp + 48);
            float4 y3 = *(const float4*)(kp + 52);
            b0[0]=(__bf16)x0.x; b0[1]=(__bf16)x0.y; b0[2]=(__bf16)x0.z; b0[3]=(__bf16)x0.w;
            b0[4]=(__bf16)x1.x; b0[5]=(__bf16)x1.y; b0[6]=(__bf16)x1.z; b0[7]=(__bf16)x1.w;
            b0[8]=(__bf16)x2.x; b0[9]=(__bf16)x2.y; b0[10]=(__bf16)x2.z; b0[11]=(__bf16)x2.w;
            b0[12]=(__bf16)x3.x; b0[13]=(__bf16)x3.y; b0[14]=(__bf16)x3.z; b0[15]=(__bf16)x3.w;
            b1[0]=(__bf16)y0.x; b1[1]=(__bf16)y0.y; b1[2]=(__bf16)y0.z; b1[3]=(__bf16)y0.w;
            b1[4]=(__bf16)y1.x; b1[5]=(__bf16)y1.y; b1[6]=(__bf16)y1.z; b1[7]=(__bf16)y1.w;
            b1[8]=(__bf16)y2.x; b1[9]=(__bf16)y2.y; b1[10]=(__bf16)y2.z; b1[11]=(__bf16)y2.w;
            b1[12]=(__bf16)y3.x; b1[13]=(__bf16)y3.y; b1[14]=(__bf16)y3.z; b1[15]=(__bf16)y3.w;
        }

        f32x8 c = {};
        c = __builtin_amdgcn_wmma_f32_16x16x32_bf16(false, a0, false, b0, (short)0, c, false, false);
        c = __builtin_amdgcn_wmma_f32_16x16x32_bf16(false, a1, false, b1, (short)0, c, false, false);

        const int col = kb * 16 + ln16;
        // prefetch next key block's bias rows (streaming 268MB -> hide latency)
        if (t < 15)
            __builtin_prefetch(&bias[maskBase + (size_t)(kh * 8) * S_ + (size_t)col + 16], 0, 0);

#pragma unroll
        for (int r = 0; r < 8; ++r) {
            const int row = r + kh * 8;
            const size_t gi = maskBase + (size_t)row * S_ + col;
            float s = c[r] * SCALE + bias[gi];
            if (mask[gi]) s = NEGV;
            probs[row * RS + col] = s;
            mx[r] = fmaxf(mx[r], s);
        }
    }

    // row-max across the 16 lanes that share a row (xor stays inside halves)
#pragma unroll
    for (int r = 0; r < 8; ++r) {
        float v = mx[r];
        v = fmaxf(v, __shfl_xor(v, 1));
        v = fmaxf(v, __shfl_xor(v, 2));
        v = fmaxf(v, __shfl_xor(v, 4));
        v = fmaxf(v, __shfl_xor(v, 8));
        mx[r] = v;
    }
    if (lane == 0) {
#pragma unroll
        for (int r = 0; r < 8; ++r) wmax[wave * QT + r] = mx[r];
    }
    if (lane == 16) {
#pragma unroll
        for (int r = 0; r < 8; ++r) wmax[wave * QT + 8 + r] = mx[r];
    }
    __syncthreads();

    // ---------------- Phase 3: softmax (4 rows per wave) ---------------------
#pragma unroll
    for (int i = 0; i < 4; ++i) {
        const int m = wave * 4 + i;
        float rm = fmaxf(fmaxf(wmax[m], wmax[QT + m]),
                         fmaxf(wmax[2 * QT + m], wmax[3 * QT + m]));
        float* prow = probs + m * RS;
        float lsum = 0.0f;
#pragma unroll
        for (int cit = 0; cit < 8; ++cit) {
            const int c0 = (cit * 32 + lane) * 4;
            float4 v = *(float4*)(prow + c0);
            v.x = __expf(v.x - rm);
            v.y = __expf(v.y - rm);
            v.z = __expf(v.z - rm);
            v.w = __expf(v.w - rm);
            *(float4*)(prow + c0) = v;
            lsum += (v.x + v.y) + (v.z + v.w);
        }
        lsum += __shfl_xor(lsum, 1);
        lsum += __shfl_xor(lsum, 2);
        lsum += __shfl_xor(lsum, 4);
        lsum += __shfl_xor(lsum, 8);
        lsum += __shfl_xor(lsum, 16);
        if (lane == 0) rowsum[m] = lsum;

        const float inv = 1.0f / lsum;
        float* arow = attn_out + maskBase + (size_t)m * S_;
#pragma unroll
        for (int cit = 0; cit < 8; ++cit) {
            const int c0 = (cit * 32 + lane) * 4;
            float4 v = *(float4*)(prow + c0);
            v.x *= inv; v.y *= inv; v.z *= inv; v.w *= inv;
            *(float4*)(arow + c0) = v;       // coalesced float4 stream-out
        }
    }
    __syncthreads();

    // ---------------- Phase 4: context = P x V (wave owns 16 output dims) ----
    const int nb = wave;                      // output d-tile
    f32x8 acc = {};
    for (int ks = 0; ks < 32; ++ks) {         // 32 chunks of 32 keys
        bf16x16 pa, pb;
        const float* par = probs + ln16 * RS + ks * 32 + kh * 8;
#pragma unroll
        for (int j = 0; j < 16; ++j)
            pa[j] = (__bf16)par[a_koff(j)];   // conflict-free: RS stride pads banks

        const float* vp = V + kvBase + (size_t)(ks * 32 + kh * 8) * DK + nb * 16 + ln16;
#pragma unroll
        for (int j = 0; j < 16; ++j)
            pb[j] = (__bf16)vp[(size_t)(j + (j >> 3) * 8) * DK];  // per-j: 16 lanes read 64B contiguous

        acc = __builtin_amdgcn_wmma_f32_16x16x32_bf16(false, pa, false, pb, (short)0, acc, false, false);
    }
#pragma unroll
    for (int r = 0; r < 8; ++r) {
        const int row = r + kh * 8;
        const float inv = 1.0f / rowsum[row];
        ctx_out[ctxBase + (size_t)row * DK + nb * 16 + ln16] = acc[r] * inv;
    }
}

extern "C" void kernel_launch(void* const* d_in, const int* in_sizes, int n_in,
                              void* d_out, int out_size, void* d_ws, size_t ws_size,
                              hipStream_t stream) {
    (void)in_sizes; (void)n_in; (void)out_size; (void)d_ws; (void)ws_size;
    const float*   Q    = (const float*)d_in[0];
    const float*   K    = (const float*)d_in[1];
    const float*   V    = (const float*)d_in[2];
    const uint8_t* mask = (const uint8_t*)d_in[3];   // jnp bool -> 1 byte
    const float*   bias = (const float*)d_in[4];

    float* ctx  = (float*)d_out;                          // B*H*S*DK floats
    float* attn = ctx + (size_t)B_ * H_ * S_ * DK;        // B*H*S*S floats

    const dim3 grid(B_ * H_ * (S_ / QT));                 // 4096 workgroups
    const dim3 block(128);                                // 4 waves (wave32)
    const size_t smem = (size_t)(QT * RS + 4 * QT + QT) * sizeof(float); // ~64.6 KB
    sdpa_wmma_kernel<<<grid, block, smem, stream>>>(Q, K, V, mask, bias, ctx, attn);
}
